// GraphConv_22505628631473
// MI455X (gfx1250) — matrix operand order repaired and checked by reading, not compiled, not run on hardware
//
#include <hip/hip_runtime.h>

// ---------------- problem constants (from reference) ----------------
#define NU 100000
#define NI 50000
#define NT 10000
#define NN (NU + NI + NT)
#define E_EDGES 800000
#define D 64

typedef __attribute__((ext_vector_type(16))) __bf16    v16bf;
typedef __attribute__((ext_vector_type(8)))  float     v8f;
typedef __attribute__((ext_vector_type(4)))  float     f32x4;
typedef __attribute__((ext_vector_type(4)))  unsigned  u32x4;

// Monotonic float <-> unsigned key (any sign; used for seg-max of h)
__device__ __forceinline__ unsigned orderKey(float f) {
  unsigned u = __float_as_uint(f);
  return (u & 0x80000000u) ? ~u : (u | 0x80000000u);
}
__device__ __forceinline__ float invOrderKey(unsigned k) {
  unsigned u = (k & 0x80000000u) ? (k & 0x7fffffffu) : ~k;
  return __uint_as_float(u);
}
#define NEG_INF_KEY  0x007FFFFFu  // orderKey(-inf)
#define POS_INF_BITS 0x7f800000u

// ---------------- init: pack embeddings / relu(offsets), 128b ----------------
__global__ __launch_bounds__(256) void k_init_nodes(
    const float* __restrict__ ue, const float* __restrict__ uo,
    const float* __restrict__ ie, const float* __restrict__ io,
    const float* __restrict__ te, const float* __restrict__ to,
    float* __restrict__ embs, float* __restrict__ off) {
  int q = blockIdx.x * 256 + threadIdx.x;          // one float4 per thread
  if (q >= NN * (D / 4)) return;
  const int idx  = q * 4;
  const int node = idx >> 6;
  f32x4 e, o;
  if (node < NU)           { e = *(const f32x4*)(ue + idx);                 o = *(const f32x4*)(uo + idx); }
  else if (node < NU + NI) { int b = idx - NU * D;        e = *(const f32x4*)(ie + b); o = *(const f32x4*)(io + b); }
  else                     { int b = idx - (NU + NI) * D; e = *(const f32x4*)(te + b); o = *(const f32x4*)(to + b); }
#pragma unroll
  for (int j = 0; j < 4; ++j) o[j] = fmaxf(o[j], 0.0f);
  *(f32x4*)(embs + idx) = e;
  *(f32x4*)(off + idx)  = o;
}

// ---------------- per-layer reset of reduction buffers, 128b ----------------
__global__ __launch_bounds__(256) void k_init_layer(
    unsigned* __restrict__ mkeys, float* __restrict__ s, float* __restrict__ agg,
    unsigned* __restrict__ bufMin, unsigned* __restrict__ bufMax) {
  int q = blockIdx.x * 256 + threadIdx.x;
  if (q >= NN * (D / 4)) return;
  const int idx = q * 4;
  *(u32x4*)(mkeys + idx)  = (u32x4){NEG_INF_KEY, NEG_INF_KEY, NEG_INF_KEY, NEG_INF_KEY};
  *(f32x4*)(s + idx)      = (f32x4){0.f, 0.f, 0.f, 0.f};
  *(f32x4*)(agg + idx)    = (f32x4){0.f, 0.f, 0.f, 0.f};
  *(u32x4*)(bufMin + idx) = (u32x4){POS_INF_BITS, POS_INF_BITS, POS_INF_BITS, POS_INF_BITS};
  *(u32x4*)(bufMax + idx) = (u32x4){0u, 0u, 0u, 0u};
}

// ---------------- edge MLP: h = relu(X W1^T + b1) W2^T + b2, bf16 WMMA ----
// One wave32 per 16-edge tile; 8 waves per block. Also does the seg-max atomic.
__global__ __launch_bounds__(256) void k_edge_mlp(
    const float* __restrict__ embs,
    const float* __restrict__ W1, const float* __restrict__ b1,
    const float* __restrict__ W2, const float* __restrict__ b2,
    const int* __restrict__ head, const int* __restrict__ tail,
    float* __restrict__ h_ws, unsigned* __restrict__ mkeys, int numTiles) {
  __shared__ __bf16 sW1[D * D];       // B source: row n = K-run of B column n
  __shared__ __bf16 sW2[D * D];
  __shared__ __bf16 sH[8][16 * D];    // per-wave relu(h1), row-major [16][64] bf16
  __shared__ float  sHF[8][16 * D];   // per-wave h2 tile for coalesced store
  __shared__ int    sHead[8][16];     // per-wave head ids of the 16 edges

  const int t = threadIdx.x;
  for (int i = t; i < D * D; i += 256) {
    sW1[i] = (__bf16)W1[i];
    sW2[i] = (__bf16)W2[i];
  }

  const int wave = t >> 5;
  const int lane = t & 31;
  const int tile = blockIdx.x * 8 + wave;
  const int M    = lane & 15;  // A row owned by this lane
  const int half = lane >> 4;  // K half (A/B) / M half (C)
  const int N    = lane & 15;  // C/D & B column owned by this lane

  const long tileBase = (long)tile * 16;
  if (lane < 16) {
    long eh = tileBase + lane;
    sHead[wave][lane] = head[(tile < numTiles && eh < (long)E_EDGES) ? eh : 0];
  }
  __syncthreads();  // sW1/sW2/sHead visible

  // --- gather A fragments (16x32 bf16, contiguous K order k = 16*half + s) ---
  long e  = tileBase + M;
  long ec = (tile < numTiles && e < (long)E_EDGES) ? e : 0;
  const int    tl = tail[ec];
  const float* xr = embs + (long)tl * D;

  v16bf a0, a1;
#pragma unroll
  for (int v = 0; v < 8; ++v) {
    const int kb = 2 * v + 16 * half;
    a0[2 * v]     = (__bf16)xr[kb];
    a0[2 * v + 1] = (__bf16)xr[kb + 1];
    a1[2 * v]     = (__bf16)xr[32 + kb];
    a1[2 * v + 1] = (__bf16)xr[32 + kb + 1];
  }

  // --- GEMM1: 4 column blocks, K=64 as two K32 WMMAs each ---
  __bf16* hrow = &sH[wave][0];
#pragma unroll
  for (int nb = 0; nb < 4; ++nb) {
    v8f acc = {};
    const v16bf bA = *(const v16bf*)&sW1[(nb * 16 + N) * D + 16 * half];
    acc = __builtin_amdgcn_wmma_f32_16x16x32_bf16(false, a0, false, bA, (short)0, acc, false, false);
    const v16bf bB = *(const v16bf*)&sW1[(nb * 16 + N) * D + 32 + 16 * half];
    acc = __builtin_amdgcn_wmma_f32_16x16x32_bf16(false, a1, false, bB, (short)0, acc, false, false);
    const float bias = b1[nb * 16 + N];
#pragma unroll
    for (int r = 0; r < 8; ++r) {  // C layout: lane holds column N, rows r + 8*half
      hrow[(r + 8 * half) * D + nb * 16 + N] = (__bf16)fmaxf(acc[r] + bias, 0.0f);
    }
  }
  __syncthreads();  // C-layout writes -> A-layout reads of sH

  // --- rebuild A fragments from relu(h1) tile ---
  const __bf16* hp = &sH[wave][M * D];
  v16bf c0, c1;
#pragma unroll
  for (int v = 0; v < 8; ++v) {
    const int kb = 2 * v + 16 * half;
    c0[2 * v]     = hp[kb];
    c0[2 * v + 1] = hp[kb + 1];
    c1[2 * v]     = hp[32 + kb];
    c1[2 * v + 1] = hp[32 + kb + 1];
  }

  // --- GEMM2: stage h tile in LDS + atomic seg-max into mkeys[head] ---
#pragma unroll
  for (int nb = 0; nb < 4; ++nb) {
    v8f acc = {};
    const v16bf bA = *(const v16bf*)&sW2[(nb * 16 + N) * D + 16 * half];
    acc = __builtin_amdgcn_wmma_f32_16x16x32_bf16(false, c0, false, bA, (short)0, acc, false, false);
    const v16bf bB = *(const v16bf*)&sW2[(nb * 16 + N) * D + 32 + 16 * half];
    acc = __builtin_amdgcn_wmma_f32_16x16x32_bf16(false, c1, false, bB, (short)0, acc, false, false);
    const float bias = b2[nb * 16 + N];
    const int   col  = nb * 16 + N;
#pragma unroll
    for (int r = 0; r < 8; ++r) {
      const int   row = r + 8 * half;
      const float hv  = acc[r] + bias;
      sHF[wave][row * D + col] = hv;
      if (tile < numTiles && (tileBase + row) < (long)E_EDGES) {
        const int hd = sHead[wave][row];
        atomicMax(&mkeys[(long)hd * D + col], orderKey(hv));
      }
    }
  }
  __syncthreads();  // h tile complete

  // --- coalesced 128b store of the contiguous 16x64 tile ---
  if (tile < numTiles) {
    float*       dst = h_ws + tileBase * D;
    const float* src = &sHF[wave][0];
#pragma unroll
    for (int i = 0; i < 8; ++i) {
      const int flat = i * 128 + lane * 4;
      *(f32x4*)(dst + flat) = *(const f32x4*)(src + flat);
    }
  }
}

// ---------------- e = exp(h - m[head]); s[head] += e  (float4 per thread) ----
__global__ __launch_bounds__(256) void k_exp_sum(
    float* __restrict__ h_ws, const int* __restrict__ head,
    const unsigned* __restrict__ mkeys, float* __restrict__ s) {
  int q = blockIdx.x * 256 + threadIdx.x;
  if (q >= E_EDGES * (D / 4)) return;
  const int e  = q >> 4;
  const int cg = (q & 15) * 4;
  const int hd = head[e];
  const u32x4 mk = *(const u32x4*)(mkeys + hd * D + cg);
  f32x4 hv = *(const f32x4*)(h_ws + e * D + cg);
  f32x4 ex;
#pragma unroll
  for (int j = 0; j < 4; ++j) ex[j] = __expf(hv[j] - invOrderKey(mk[j]));
  *(f32x4*)(h_ws + e * D + cg) = ex;
#pragma unroll
  for (int j = 0; j < 4; ++j) atomicAdd(&s[hd * D + cg + j], ex[j]);
}

// ------- fused: agg[head] += (e/s[head])*embs[tail]  AND offset min/max -------
__global__ __launch_bounds__(256) void k_agg_off(
    const float* __restrict__ h_ws, const int* __restrict__ head,
    const int* __restrict__ tail, const float* __restrict__ s,
    const float* __restrict__ embs, float* __restrict__ agg,
    const float* __restrict__ off, unsigned* __restrict__ bufMin,
    unsigned* __restrict__ bufMax) {
  int q = blockIdx.x * 256 + threadIdx.x;
  if (q >= E_EDGES * (D / 4)) return;
  const int e  = q >> 4;
  const int cg = (q & 15) * 4;
  const int hd = head[e], tl = tail[e];

  // attention-weighted aggregation
  const f32x4 ev = *(const f32x4*)(h_ws + e * D + cg);
  const f32x4 sv = *(const f32x4*)(s + hd * D + cg);
  const f32x4 xv = *(const f32x4*)(embs + (long)tl * D + cg);
#pragma unroll
  for (int j = 0; j < 4; ++j) atomicAdd(&agg[hd * D + cg + j], (ev[j] / sv[j]) * xv[j]);

  // masked seg-min/seg-max of relu(off[tail])  (nonneg: uint order == float order)
  const f32x4 ov = *(const f32x4*)(off + (long)tl * D + cg);
  u32x4 b;
#pragma unroll
  for (int j = 0; j < 4; ++j) b[j] = __float_as_uint(fmaxf(ov[j], 0.0f));
  if (hd < NU) {
    if (tl >= NU && tl < NU + NI) {          // inter_mask -> seg_min
#pragma unroll
      for (int j = 0; j < 4; ++j) atomicMin(&bufMin[hd * D + cg + j], b[j]);
    } else if (tl >= NU + NI) {              // ut_mask -> seg_max
#pragma unroll
      for (int j = 0; j < 4; ++j) atomicMax(&bufMax[hd * D + cg + j], b[j]);
    }
  } else if (hd < NU + NI) {                 // item_mask -> seg_max
#pragma unroll
    for (int j = 0; j < 4; ++j) atomicMax(&bufMax[hd * D + cg + j], b[j]);
  } else {                                   // tag_mask -> seg_min
#pragma unroll
    for (int j = 0; j < 4; ++j) atomicMin(&bufMin[hd * D + cg + j], b[j]);
  }
}

// ---------------- normalize agg rows; combine offset buffers ----------------
__global__ __launch_bounds__(256) void k_node_update(
    const float* __restrict__ agg, const unsigned* __restrict__ bufMin,
    const unsigned* __restrict__ bufMax, float* __restrict__ embs,
    float* __restrict__ off) {
  int n = blockIdx.x * 256 + threadIdx.x;
  if (n >= NN) return;
  float ss = 0.0f;
#pragma unroll
  for (int cq = 0; cq < D / 4; ++cq) {
    const f32x4 a = *(const f32x4*)(agg + n * D + cq * 4);
    ss += a[0] * a[0] + a[1] * a[1] + a[2] * a[2] + a[3] * a[3];
  }
  const float inv = 1.0f / fmaxf(sqrtf(ss), 1e-12f);
#pragma unroll
  for (int cq = 0; cq < D / 4; ++cq) {
    const int base = n * D + cq * 4;
    const f32x4 a  = *(const f32x4*)(agg + base);
    const u32x4 km = *(const u32x4*)(bufMin + base);
    const u32x4 kM = *(const u32x4*)(bufMax + base);
    f32x4 eo, oo;
#pragma unroll
    for (int j = 0; j < 4; ++j) {
      eo[j] = a[j] * inv;
      const float vmin = (km[j] == POS_INF_BITS) ? 0.0f : __uint_as_float(km[j]); // finite_or_zero
      const float vmax = __uint_as_float(kM[j]);  // seed 0 == finite_or_zero(-inf) for relu'd vals
      float o;
      if (n < NU)           o = fminf(vmin, vmax);  // relu(min(inter,ut)), both >= 0
      else if (n < NU + NI) o = vmax;
      else                  o = vmin;
      oo[j] = fmaxf(o, 0.0f);
    }
    *(f32x4*)(embs + base) = eo;
    *(f32x4*)(off + base)  = oo;
  }
}

// ---------------- final layout: (ue, uo, ie, io, te, to) flat ----------------
__global__ __launch_bounds__(256) void k_writeout(
    const float* __restrict__ embs, const float* __restrict__ off,
    float* __restrict__ out) {
  int q = blockIdx.x * 256 + threadIdx.x;
  if (q >= NN * (D / 4)) return;
  const int idx  = q * 4;
  const int node = idx >> 6;
  long eo, oo;
  if (node < NU) {
    eo = idx;                         oo = (long)NU * D + idx;
  } else if (node < NU + NI) {
    const long b = idx - (long)NU * D;
    eo = 2L * NU * D + b;             oo = 2L * NU * D + (long)NI * D + b;
  } else {
    const long b = idx - (long)(NU + NI) * D;
    eo = 2L * (NU + NI) * D + b;      oo = 2L * (NU + NI) * D + (long)NT * D + b;
  }
  *(f32x4*)(out + eo) = *(const f32x4*)(embs + idx);
  *(f32x4*)(out + oo) = *(const f32x4*)(off + idx);
}

extern "C" void kernel_launch(void* const* d_in, const int* in_sizes, int n_in,
                              void* d_out, int out_size, void* d_ws, size_t ws_size,
                              hipStream_t stream) {
  const float* ue = (const float*)d_in[0];
  const float* uo = (const float*)d_in[1];
  const float* ie = (const float*)d_in[2];
  const float* io = (const float*)d_in[3];
  const float* te = (const float*)d_in[4];
  const float* to = (const float*)d_in[5];
  const float* W1 = (const float*)d_in[6];
  const float* b1 = (const float*)d_in[7];
  const float* W2 = (const float*)d_in[8];
  const float* b2 = (const float*)d_in[9];
  const int* head = (const int*)d_in[10];
  const int* tail = (const int*)d_in[11];
  float* out = (float*)d_out;

  // workspace carve-up (regions 256B-aligned by construction)
  char* p = (char*)d_ws;
  float*    h_ws   = (float*)p;    p += (size_t)E_EDGES * D * 4;  // 204.8 MB
  unsigned* mkeys  = (unsigned*)p; p += (size_t)NN * D * 4;
  float*    s      = (float*)p;    p += (size_t)NN * D * 4;
  float*    agg    = (float*)p;    p += (size_t)NN * D * 4;
  unsigned* bufMin = (unsigned*)p; p += (size_t)NN * D * 4;
  unsigned* bufMax = (unsigned*)p; p += (size_t)NN * D * 4;
  float*    embs   = (float*)p;    p += (size_t)NN * D * 4;
  float*    off    = (float*)p;    p += (size_t)NN * D * 4;

  const int NDQ = NN * (D / 4);        // float4 granules per node pass
  const int EDQ = E_EDGES * (D / 4);   // float4 granules per edge pass
  const int numTiles = (E_EDGES + 15) / 16;
  dim3 blk(256);

  k_init_nodes<<<(NDQ + 255) / 256, blk, 0, stream>>>(ue, uo, ie, io, te, to, embs, off);

  for (int layer = 0; layer < 2; ++layer) {
    k_init_layer<<<(NDQ + 255) / 256, blk, 0, stream>>>(mkeys, s, agg, bufMin, bufMax);
    k_edge_mlp<<<(numTiles + 7) / 8, blk, 0, stream>>>(embs, W1, b1, W2, b2, head, tail,
                                                       h_ws, mkeys, numTiles);
    k_exp_sum<<<(EDQ + 255) / 256, blk, 0, stream>>>(h_ws, head, mkeys, s);
    k_agg_off<<<(EDQ + 255) / 256, blk, 0, stream>>>(h_ws, head, tail, s, embs, agg,
                                                     off, bufMin, bufMax);
    k_node_update<<<(NN + 255) / 256, blk, 0, stream>>>(agg, bufMin, bufMax, embs, off);
  }

  k_writeout<<<(NDQ + 255) / 256, blk, 0, stream>>>(embs, off, out);
}
